// ParticleNet_53712861003951
// MI455X (gfx1250) — matrix operand order repaired and checked by reading, not compiled, not run on hardware
//
#include <hip/hip_runtime.h>
#include <hip/hip_bf16.h>

// ---------------------------------------------------------------------------
// ParticleNet forward for MI455X (gfx1250, wave32, WMMA).
// - All GEMM stages: v_wmma_f32_16x16x32_bf16 (bf16 A/B, f32 acc).
// - Edge-conv activations kept as bf16 edge-major so B tiles are two 128-bit
//   loads (no per-element converts) -> halves HBM traffic of the dominant
//   stage (23.3 TB/s roofline, workload is bandwidth-bound).
// - KNN2 Gram loop double-buffers candidate rows through LDS with
//   global_load_async_to_lds_b128 + s_wait_asynccnt (ASYNCcnt path).
// ---------------------------------------------------------------------------

#define B_    128
#define P_    1024
#define KNN_  7
#define CIN_  7
#define NCLS_ 10
#define PK_   (P_ * KNN_)   // 7168 edges per batch
#define BIGF  1000000000.0f
#define CB_   16            // batches per edge-conv chunk

typedef __attribute__((ext_vector_type(16))) __bf16       v16bf;
typedef __attribute__((ext_vector_type(8)))  float        v8f;
typedef __attribute__((ext_vector_type(4)))  unsigned int u32x4;

static __device__ __forceinline__ unsigned short f2bf_bits(float f) {
  __bf16 h = (__bf16)f;
  return __builtin_bit_cast(unsigned short, h);
}
static __device__ __forceinline__ float bf2f(unsigned short u) {
  unsigned v = ((unsigned)u) << 16;
  return __builtin_bit_cast(float, v);
}

// ---------------------------------------------------------------------------
// Weight pre-layout: f32 (Oreal x Creal) -> bf16 tiles in a contiguous-K 16-bit
// A-matrix (16x32) operand order (matching the B tiles) for V_WMMA_F32_16X16X32_BF16:
//   lane L: M = L%16 ; half h:
//   K = 16*(L>=16) + h
// Stored as Aw[((mt*KT+kt)*32 + lane)*16 + h] -> two 128-bit loads per tile.
// ---------------------------------------------------------------------------
__global__ void pn_prep_w(const float* __restrict__ W, unsigned short* __restrict__ Aw,
                          int Oreal, int Creal, int MT, int KT) {
  int t = blockIdx.x * blockDim.x + threadIdx.x;
  int total = MT * KT * 32;
  if (t >= total) return;
  int lane = t & 31;
  int tile = t >> 5;
  int kt = tile % KT;
  int mt = tile / KT;
  int m  = mt * 16 + (lane & 15);
  int hi = lane >> 4;
  for (int h = 0; h < 16; ++h) {
    int v = h >> 1, j = h & 1;
    int k = hi * 16 + h;
    int c = kt * 32 + k;
    float val = (m < Oreal && c < Creal) ? W[m * Creal + c] : 0.f;
    Aw[(size_t)t * 16 + h] = f2bf_bits(val);
  }
}

// ---------------------------------------------------------------------------
// WMMA conv over f32 channel-major input -> f32 channel-major output.
// Used for shortcut/fusion stages (BN, optional ReLU, optional mask).
// ---------------------------------------------------------------------------
template <int MT, int KT, int RELU, int USE_MASK>
__global__ void __launch_bounds__(32)
pn_wmma_conv(const unsigned short* __restrict__ Aw,
             const float* __restrict__ X, float* __restrict__ Y,
             const float* __restrict__ bnp,     // [4][Oreal]
             const float* __restrict__ mask,    // [b][N] or nullptr
             int Creal, int Oreal, int N) {
  const int lane = threadIdx.x & 31;
  const int hi   = lane >> 4;
  const int nt   = blockIdx.x;
  const int b    = blockIdx.y;
  const int n    = nt * 16 + (lane & 15);
  const float* Xb = X + (size_t)b * Creal * N;

  v8f acc[MT];
#pragma unroll
  for (int mt = 0; mt < MT; ++mt) acc[mt] = (v8f){};

#pragma unroll
  for (int kt = 0; kt < KT; ++kt) {
    v16bf bt;
    const int kbase = kt * 32 + hi * 16;
#pragma unroll
    for (int h = 0; h < 16; ++h) {
      const int c = kbase + h;
      float xv = (c < Creal) ? Xb[(size_t)c * N + n] : 0.f;
      bt[h] = (__bf16)xv;
    }
#pragma unroll
    for (int mt = 0; mt < MT; ++mt) {
      const u32x4* ap = (const u32x4*)(Aw + ((size_t)(mt * KT + kt) * 32 + lane) * 16);
      union { u32x4 u[2]; v16bf v; } at;
      at.u[0] = ap[0];
      at.u[1] = ap[1];
      acc[mt] = __builtin_amdgcn_wmma_f32_16x16x32_bf16(
          false, at.v, false, bt, (short)0, acc[mt], false, false);
    }
  }

  float mval = 1.f;
  if (USE_MASK) mval = mask[(size_t)b * N + n];
#pragma unroll
  for (int mt = 0; mt < MT; ++mt) {
#pragma unroll
    for (int v = 0; v < 8; ++v) {
      const int o = mt * 16 + v + hi * 8;
      if (o < Oreal) {
        float g  = bnp[o];
        float be = bnp[Oreal + o];
        float mu = bnp[2 * Oreal + o];
        float va = bnp[3 * Oreal + o];
        float s  = g * rsqrtf(va + 1e-5f);
        float val = (acc[mt][v] - mu) * s + be;
        if (RELU) val = fmaxf(val, 0.f);
        if (USE_MASK) val *= mval;
        Y[((size_t)b * Oreal + o) * N + n] = val;
      }
    }
  }
}

// ---------------------------------------------------------------------------
// WMMA conv over bf16 EDGE-MAJOR data: X16[b][n][CPAD] -> Y16[b][n][MT*16],
// BN+ReLU fused, bf16 out. B tile = two 128-bit row loads (halves h map to
// K = 16*(lane>=16)+h which is contiguous in the row). Epilogue packs the 8
// consecutive output channels per lane into one 16-byte store.
// ---------------------------------------------------------------------------
template <int MT, int KT, int CPAD>
__global__ void __launch_bounds__(32)
pn_wmma_conv16(const unsigned short* __restrict__ Aw,
               const unsigned short* __restrict__ X16,
               unsigned short* __restrict__ Y16,
               const float* __restrict__ bnp,  // [4][MT*16]
               int N) {
  const int Oreal = MT * 16;
  const int lane = threadIdx.x & 31;
  const int hi   = lane >> 4;
  const int nt   = blockIdx.x;
  const int b    = blockIdx.y;
  const int n    = nt * 16 + (lane & 15);
  const unsigned short* row = X16 + ((size_t)b * N + n) * CPAD;

  v8f acc[MT];
#pragma unroll
  for (int mt = 0; mt < MT; ++mt) acc[mt] = (v8f){};

#pragma unroll
  for (int kt = 0; kt < KT; ++kt) {
    union { u32x4 u[2]; v16bf v; } bt;
    const u32x4* rp = (const u32x4*)(row + kt * 32 + hi * 16);
    bt.u[0] = rp[0];
    bt.u[1] = rp[1];
#pragma unroll
    for (int mt = 0; mt < MT; ++mt) {
      const u32x4* ap = (const u32x4*)(Aw + ((size_t)(mt * KT + kt) * 32 + lane) * 16);
      union { u32x4 u[2]; v16bf v; } at;
      at.u[0] = ap[0];
      at.u[1] = ap[1];
      acc[mt] = __builtin_amdgcn_wmma_f32_16x16x32_bf16(
          false, at.v, false, bt.v, (short)0, acc[mt], false, false);
    }
  }

  unsigned short* Yrow = Y16 + ((size_t)b * N + n) * Oreal;
#pragma unroll
  for (int mt = 0; mt < MT; ++mt) {
    float f[8];
#pragma unroll
    for (int v = 0; v < 8; ++v) {
      const int o = mt * 16 + hi * 8 + v;  // consecutive channels
      float g  = bnp[o];
      float be = bnp[Oreal + o];
      float mu = bnp[2 * Oreal + o];
      float va = bnp[3 * Oreal + o];
      float s  = g * rsqrtf(va + 1e-5f);
      f[v] = fmaxf((acc[mt][v] - mu) * s + be, 0.f);
    }
    u32x4 pk;
#pragma unroll
    for (int w = 0; w < 4; ++w)
      pk[w] = (unsigned)f2bf_bits(f[2 * w]) | ((unsigned)f2bf_bits(f[2 * w + 1]) << 16);
    *(u32x4*)(Yrow + mt * 16 + hi * 8) = pk;
  }
}

// ---------------------------------------------------------------------------
// KNN over 32-dim features via WMMA Gram tiles with async LDS double-buffer.
// Xt: point-major bf16 [b][p][32] (64B rows); nrm[b][p] = sum_c y^2 (f32).
// Ranking key 2*q.c - ||c||^2 (row-constant -||q||^2 dropped). Self excluded
// by index; top-7 via register bubble list in lanes 0..15.
// ---------------------------------------------------------------------------
__global__ void __launch_bounds__(32)
pn_knn_wmma(const unsigned short* __restrict__ Xt, const float* __restrict__ nrm,
            int* __restrict__ idxOut) {
  __shared__ float pdl[16][17];
  __shared__ unsigned short bbuf[2][16 * 32];  // 2 stages x 16 rows x 64B

  const int lane = threadIdx.x & 31;
  const int hi   = lane >> 4;
  const int r    = lane & 15;
  const int ch   = hi;           // 16B chunk id for async copy
  const int b    = blockIdx.y;
  const int qt   = blockIdx.x;
  const int q    = qt * 16 + r;
  const unsigned short* Xb = Xt + (size_t)b * P_ * 32;

  // A tile: halves 0..15 = K [16*hi..16*hi+15]
  const u32x4* arow = (const u32x4*)(Xb + (size_t)q * 32);
  union { u32x4 u[2]; v16bf v; } at, bt;
  at.u[0] = arow[2 * hi];
  at.u[1] = arow[2 * hi + 1];

  // Async copy of one 16x64B candidate-row tile into bbuf[stage]:
  // each lane moves two 16B chunks (rows r, chunks ch and ch+2).
  auto issue = [&](int ct, int stage) {
    const unsigned short* g0 = Xb + (size_t)(ct * 16 + r) * 32 + ch * 8;
    unsigned l0 = (unsigned)(unsigned long long)(void*)&bbuf[stage][r * 32 + ch * 8];
    asm volatile("global_load_async_to_lds_b128 %0, %1, off"
                 :: "v"(l0), "v"((unsigned long long)g0) : "memory");
    asm volatile("global_load_async_to_lds_b128 %0, %1, off"
                 :: "v"(l0 + 32u), "v"((unsigned long long)(g0 + 16)) : "memory");
  };

  float bd[KNN_];
  int   bi[KNN_];
#pragma unroll
  for (int j = 0; j < KNN_; ++j) { bd[j] = -3.0e38f; bi[j] = 0; }

  issue(0, 0);
  for (int ct = 0; ct < P_ / 16; ++ct) {
    if (ct + 1 < P_ / 16) {
      issue(ct + 1, (ct + 1) & 1);
      // current tile's 2 async loads are older than the 2 just issued:
      asm volatile("s_wait_asynccnt 0x2" ::: "memory");
    } else {
      asm volatile("s_wait_asynccnt 0x0" ::: "memory");
    }
    const u32x4* rp = (const u32x4*)&bbuf[ct & 1][r * 32 + hi * 16];
    bt.u[0] = rp[0];
    bt.u[1] = rp[1];

    v8f z = (v8f){};
    v8f acc = __builtin_amdgcn_wmma_f32_16x16x32_bf16(
        false, at.v, false, bt.v, (short)0, z, false, false);

    const int cn = ct * 16 + r;
    const float nc = nrm[(size_t)b * P_ + cn];
#pragma unroll
    for (int v = 0; v < 8; ++v)
      pdl[v + hi * 8][r] = 2.f * acc[v] - nc;
    __syncthreads();

    if (lane < 16) {
#pragma unroll
      for (int nn = 0; nn < 16; ++nn) {
        int c = ct * 16 + nn;
        if (c == q) continue;
        float val = pdl[lane][nn];
        int ci = c;
#pragma unroll
        for (int j = 0; j < KNN_; ++j) {
          if (val > bd[j]) {
            float tv = bd[j]; int ti = bi[j];
            bd[j] = val; bi[j] = ci;
            val = tv; ci = ti;
          }
        }
      }
    }
    __syncthreads();
  }

  if (lane < 16) {
#pragma unroll
    for (int j = 0; j < KNN_; ++j)
      idxOut[((size_t)b * P_ + q) * KNN_ + j] = bi[j];
  }
}

// ---------------------------------------------------------------------------
// KNN over 2-D coordinates (K=2 too small for WMMA): scalar with LDS-staged
// points, one block per batch.
// ---------------------------------------------------------------------------
__global__ void pn_knn_small(const float* __restrict__ pts, int* __restrict__ idxOut) {
  __shared__ float px[P_], py[P_];
  const int b = blockIdx.x;
  const float* pb = pts + (size_t)b * 2 * P_;
  for (int i = threadIdx.x; i < P_; i += blockDim.x) {
    px[i] = pb[i];
    py[i] = pb[P_ + i];
  }
  __syncthreads();
  for (int q = threadIdx.x; q < P_; q += blockDim.x) {
    float qx = px[q], qy = py[q];
    float bd[KNN_];
    int   bi[KNN_];
#pragma unroll
    for (int j = 0; j < KNN_; ++j) { bd[j] = -3.0e38f; bi[j] = 0; }
    for (int c = 0; c < P_; ++c) {
      if (c == q) continue;
      float dx = px[c] - qx, dy = py[c] - qy;
      float val = -(dx * dx + dy * dy);
      int ci = c;
#pragma unroll
      for (int j = 0; j < KNN_; ++j) {
        if (val > bd[j]) {
          float tv = bd[j]; int ti = bi[j];
          bd[j] = val; bi[j] = ci;
          val = tv; ci = ti;
        }
      }
    }
#pragma unroll
    for (int j = 0; j < KNN_; ++j)
      idxOut[((size_t)b * P_ + q) * KNN_ + j] = bi[j];
  }
}

// ---------------------------------------------------------------------------
// Elementwise stages
// ---------------------------------------------------------------------------
__global__ void pn_preprocess(const float* __restrict__ points,
                              const float* __restrict__ features,
                              const float* __restrict__ mask,
                              const float* __restrict__ bnf,
                              float* __restrict__ ptsS, float* __restrict__ ftsb) {
  size_t t = (size_t)blockIdx.x * blockDim.x + threadIdx.x;
  if (t >= (size_t)B_ * P_) return;
  int p = (int)(t % P_);
  int b = (int)(t / P_);
  float m = mask[t];
  float shift = (m == 0.f) ? BIGF : 0.f;
  ptsS[(size_t)b * 2 * P_ + p]      = points[(size_t)b * 2 * P_ + p] * m + shift;
  ptsS[(size_t)b * 2 * P_ + P_ + p] = points[(size_t)b * 2 * P_ + P_ + p] * m + shift;
#pragma unroll
  for (int c = 0; c < CIN_; ++c) {
    float f  = features[((size_t)b * CIN_ + c) * P_ + p] * m;
    float g  = bnf[c];
    float be = bnf[CIN_ + c];
    float mu = bnf[2 * CIN_ + c];
    float va = bnf[3 * CIN_ + c];
    ftsb[((size_t)b * CIN_ + c) * P_ + p] = ((f - mu) * rsqrtf(va + 1e-5f) * g + be) * m;
  }
}

__global__ void pn_counts(const float* __restrict__ mask, float* __restrict__ counts) {
  __shared__ float red[256];
  const int b = blockIdx.x;
  float s = 0.f;
  for (int p = threadIdx.x; p < P_; p += 256) s += mask[(size_t)b * P_ + p];
  red[threadIdx.x] = s;
  __syncthreads();
  for (int w = 128; w > 0; w >>= 1) {
    if ((int)threadIdx.x < w) red[threadIdx.x] += red[threadIdx.x + w];
    __syncthreads();
  }
  if (threadIdx.x == 0) counts[b] = fmaxf(red[0], 1.f);
}

// Edge features, bf16 edge-major: ef[b][n][c], n = p*K+k, row padded to Cpad.
__global__ void pn_edge_feat16(const float* __restrict__ x, const int* __restrict__ idx,
                               unsigned short* __restrict__ ef, int C, int Cpad, int nb) {
  size_t t = (size_t)blockIdx.x * blockDim.x + threadIdx.x;
  size_t total = (size_t)nb * PK_ * Cpad;
  if (t >= total) return;
  int c = (int)(t % Cpad);
  size_t r = t / Cpad;
  int n = (int)(r % PK_);
  int b = (int)(r / PK_);
  int p = n / KNN_;
  int k = n % KNN_;
  const float* xb = x + (size_t)b * C * P_;
  float val = 0.f;
  if (c < C) {
    val = xb[(size_t)c * P_ + p];
  } else if (c < 2 * C) {
    int nbi = idx[((size_t)b * P_ + p) * KNN_ + k];
    int cc = c - C;
    val = xb[(size_t)cc * P_ + nbi] - xb[(size_t)cc * P_ + p];
  }
  ef[t] = f2bf_bits(val);
}

// Mean over k of bf16 edge-major conv output -> f32 channel-major.
__global__ void pn_mean16(const unsigned short* __restrict__ l3, float* __restrict__ outm,
                          int O, int nb) {
  size_t t = (size_t)blockIdx.x * blockDim.x + threadIdx.x;
  size_t total = (size_t)nb * O * P_;
  if (t >= total) return;
  int p = (int)(t % P_);
  size_t r = t / P_;
  int o = (int)(r % O);
  int b = (int)(r / O);
  const unsigned short* base = l3 + ((size_t)b * PK_ + (size_t)p * KNN_) * O + o;
  float s = 0.f;
#pragma unroll
  for (int k = 0; k < KNN_; ++k) s += bf2f(base[(size_t)k * O]);
  outm[t] = s * (1.f / (float)KNN_);
}

__global__ void pn_combine(const float* __restrict__ sc, const float* __restrict__ mn,
                           const float* __restrict__ mask, float* __restrict__ out, int O) {
  size_t t = (size_t)blockIdx.x * blockDim.x + threadIdx.x;
  size_t total = (size_t)B_ * O * P_;
  if (t >= total) return;
  int p = (int)(t % P_);
  size_t r = t / P_;
  int b = (int)(r / O);
  float m = mask[(size_t)b * P_ + p];
  out[t] = fmaxf(sc[t] + mn[t], 0.f) * m;
}

// Point-major bf16 copy of (fts1 + coord_shift) + f32 norms for KNN2.
__global__ void pn_knn2prep(const float* __restrict__ fts1, const float* __restrict__ mask,
                            unsigned short* __restrict__ Xt, float* __restrict__ nrm) {
  size_t t = (size_t)blockIdx.x * blockDim.x + threadIdx.x;
  if (t >= (size_t)B_ * P_) return;
  int p = (int)(t % P_);
  int b = (int)(t / P_);
  float shift = (mask[t] == 0.f) ? BIGF : 0.f;
  float s = 0.f;
#pragma unroll
  for (int c = 0; c < 32; ++c) {
    float y = fts1[((size_t)b * 32 + c) * P_ + p] + shift;
    s += y * y;
    Xt[t * 32 + c] = f2bf_bits(y);
  }
  nrm[t] = s;
}

__global__ void pn_concat(const float* __restrict__ a, const float* __restrict__ c2,
                          float* __restrict__ out) {
  size_t t = (size_t)blockIdx.x * blockDim.x + threadIdx.x;
  size_t total = (size_t)B_ * 96 * P_;
  if (t >= total) return;
  int p = (int)(t % P_);
  size_t r = t / P_;
  int c = (int)(r % 96);
  int b = (int)(r / 96);
  out[t] = (c < 32) ? a[((size_t)b * 32 + c) * P_ + p]
                    : c2[((size_t)b * 64 + (c - 32)) * P_ + p];
}

__global__ void pn_pool(const float* __restrict__ fused, const float* __restrict__ counts,
                        float* __restrict__ pooled) {
  const int b = blockIdx.x;
  const int o = threadIdx.x;  // 128
  const float* f = fused + ((size_t)b * 128 + o) * P_;
  float s = 0.f;
  for (int p = 0; p < P_; ++p) s += f[p];
  pooled[(size_t)b * 128 + o] = s / counts[b];
}

__global__ void pn_fc(const float* __restrict__ pooled,
                      const float* __restrict__ w1, const float* __restrict__ b1,
                      const float* __restrict__ w2, const float* __restrict__ b2,
                      float* __restrict__ out) {
  __shared__ float xv[128], h[128];
  const int b = blockIdx.x;
  const int o = threadIdx.x;  // 128
  xv[o] = pooled[(size_t)b * 128 + o];
  __syncthreads();
  float s = b1[o];
  for (int c = 0; c < 128; ++c) s += w1[o * 128 + c] * xv[c];
  h[o] = fmaxf(s, 0.f);
  __syncthreads();
  if (o < NCLS_) {
    float s2 = b2[o];
    for (int c = 0; c < 128; ++c) s2 += w2[o * 128 + c] * h[c];
    out[(size_t)b * NCLS_ + o] = s2;
  }
}

// ---------------------------------------------------------------------------
// Host orchestration
// ---------------------------------------------------------------------------
#define CEILDIV(a, b) (((a) + (b) - 1) / (b))

extern "C" void kernel_launch(void* const* d_in, const int* in_sizes, int n_in,
                              void* d_out, int out_size, void* d_ws, size_t ws_size,
                              hipStream_t stream) {
  (void)in_sizes; (void)n_in; (void)out_size; (void)ws_size;

  const float* points   = (const float*)d_in[0];
  const float* features = (const float*)d_in[1];
  const float* mask     = (const float*)d_in[2];
  const float* bn_fts   = (const float*)d_in[3];
  const float* ec1_w1   = (const float*)d_in[4];
  const float* ec1_w2   = (const float*)d_in[5];
  const float* ec1_w3   = (const float*)d_in[6];
  const float* ec1_bn1  = (const float*)d_in[7];
  const float* ec1_bn2  = (const float*)d_in[8];
  const float* ec1_bn3  = (const float*)d_in[9];
  const float* ec1_scw  = (const float*)d_in[10];
  const float* ec1_scbn = (const float*)d_in[11];
  const float* ec2_w1   = (const float*)d_in[12];
  const float* ec2_w2   = (const float*)d_in[13];
  const float* ec2_w3   = (const float*)d_in[14];
  const float* ec2_bn1  = (const float*)d_in[15];
  const float* ec2_bn2  = (const float*)d_in[16];
  const float* ec2_bn3  = (const float*)d_in[17];
  const float* ec2_scw  = (const float*)d_in[18];
  const float* ec2_scbn = (const float*)d_in[19];
  const float* fus_w    = (const float*)d_in[20];
  const float* fus_bn   = (const float*)d_in[21];
  const float* fc1_w    = (const float*)d_in[22];
  const float* fc1_b    = (const float*)d_in[23];
  const float* fc2_w    = (const float*)d_in[24];
  const float* fc2_b    = (const float*)d_in[25];
  float* out = (float*)d_out;

  // -------- workspace bump allocator (total ~250 MB) --------
  char* base = (char*)d_ws;
  size_t off = 0;
  auto alloc = [&](size_t bytes) -> char* {
    char* p = base + off;
    off += (bytes + 255) & ~(size_t)255;
    return p;
  };
  float* ptsS   = (float*)alloc((size_t)B_ * 2 * P_ * 4);
  float* ftsb   = (float*)alloc((size_t)B_ * CIN_ * P_ * 4);
  float* counts = (float*)alloc((size_t)B_ * 4);
  int*   idx1   = (int*)alloc((size_t)B_ * P_ * KNN_ * 4);
  int*   idx2   = (int*)alloc((size_t)B_ * P_ * KNN_ * 4);
  float* mean1  = (float*)alloc((size_t)B_ * 32 * P_ * 4);
  float* sc1    = (float*)alloc((size_t)B_ * 32 * P_ * 4);
  float* fts1   = (float*)alloc((size_t)B_ * 32 * P_ * 4);
  unsigned short* Xt2 = (unsigned short*)alloc((size_t)B_ * P_ * 32 * 2);
  float* nrm2   = (float*)alloc((size_t)B_ * P_ * 4);
  float* mean2  = (float*)alloc((size_t)B_ * 64 * P_ * 4);
  float* sc2    = (float*)alloc((size_t)B_ * 64 * P_ * 4);
  float* fts2   = (float*)alloc((size_t)B_ * 64 * P_ * 4);
  float* pooled = (float*)alloc((size_t)B_ * 128 * 4);

  unsigned short* awE1W1 = (unsigned short*)alloc(2 * 1 * 32 * 16 * 2);
  unsigned short* awE1W2 = (unsigned short*)alloc(2 * 1 * 32 * 16 * 2);
  unsigned short* awE1W3 = (unsigned short*)alloc(2 * 1 * 32 * 16 * 2);
  unsigned short* awE1SC = (unsigned short*)alloc(2 * 1 * 32 * 16 * 2);
  unsigned short* awE2W1 = (unsigned short*)alloc(4 * 2 * 32 * 16 * 2);
  unsigned short* awE2W2 = (unsigned short*)alloc(4 * 2 * 32 * 16 * 2);
  unsigned short* awE2W3 = (unsigned short*)alloc(4 * 2 * 32 * 16 * 2);
  unsigned short* awE2SC = (unsigned short*)alloc(4 * 1 * 32 * 16 * 2);
  unsigned short* awFUS  = (unsigned short*)alloc(8 * 3 * 32 * 16 * 2);

  // Aliased big region: bf16 edge-conv chunk buffers, later xcat+fused.
  const size_t EFB = (size_t)CB_ * PK_ * 64 * 2;  // 14,680,064 B per buffer
  char* big = alloc(117440512);
  unsigned short* ef16   = (unsigned short*)(big);
  unsigned short* bufA16 = (unsigned short*)(big + EFB);
  unsigned short* bufB16 = (unsigned short*)(big + 2 * EFB);
  float* xcat  = (float*)(big);                            // reuse after edge convs
  float* fused = (float*)(big + (size_t)B_ * 96 * P_ * 4); // 50,331,648 offset

  // -------- weight pre-layout --------
  pn_prep_w<<<1, 256, 0, stream>>>(ec1_w1, awE1W1, 32, 14, 2, 1);
  pn_prep_w<<<1, 256, 0, stream>>>(ec1_w2, awE1W2, 32, 32, 2, 1);
  pn_prep_w<<<1, 256, 0, stream>>>(ec1_w3, awE1W3, 32, 32, 2, 1);
  pn_prep_w<<<1, 256, 0, stream>>>(ec1_scw, awE1SC, 32, 7, 2, 1);
  pn_prep_w<<<1, 256, 0, stream>>>(ec2_w1, awE2W1, 64, 64, 4, 2);
  pn_prep_w<<<1, 256, 0, stream>>>(ec2_w2, awE2W2, 64, 64, 4, 2);
  pn_prep_w<<<1, 256, 0, stream>>>(ec2_w3, awE2W3, 64, 64, 4, 2);
  pn_prep_w<<<1, 256, 0, stream>>>(ec2_scw, awE2SC, 64, 32, 4, 1);
  pn_prep_w<<<CEILDIV(8 * 3 * 32, 256), 256, 0, stream>>>(fus_w, awFUS, 128, 96, 8, 3);

  // -------- preprocess, counts, KNN1 --------
  pn_preprocess<<<CEILDIV(B_ * P_, 256), 256, 0, stream>>>(points, features, mask, bn_fts, ptsS, ftsb);
  pn_counts<<<B_, 256, 0, stream>>>(mask, counts);
  pn_knn_small<<<B_, 256, 0, stream>>>(ptsS, idx1);

  // -------- EdgeConv 1 (chunked over batches, bf16 edge-major) --------
  for (int c0 = 0; c0 < B_; c0 += CB_) {
    pn_edge_feat16<<<CEILDIV((size_t)CB_ * PK_ * 32, 256), 256, 0, stream>>>(
        ftsb + (size_t)c0 * CIN_ * P_, idx1 + (size_t)c0 * P_ * KNN_, ef16, CIN_, 32, CB_);
    pn_wmma_conv16<2, 1, 32><<<dim3(PK_ / 16, CB_), 32, 0, stream>>>(
        awE1W1, ef16, bufA16, ec1_bn1, PK_);
    pn_wmma_conv16<2, 1, 32><<<dim3(PK_ / 16, CB_), 32, 0, stream>>>(
        awE1W2, bufA16, bufB16, ec1_bn2, PK_);
    pn_wmma_conv16<2, 1, 32><<<dim3(PK_ / 16, CB_), 32, 0, stream>>>(
        awE1W3, bufB16, bufA16, ec1_bn3, PK_);
    pn_mean16<<<CEILDIV((size_t)CB_ * 32 * P_, 256), 256, 0, stream>>>(
        bufA16, mean1 + (size_t)c0 * 32 * P_, 32, CB_);
  }
  pn_wmma_conv<2, 1, 0, 0><<<dim3(P_ / 16, B_), 32, 0, stream>>>(
      awE1SC, ftsb, sc1, ec1_scbn, nullptr, 7, 32, P_);
  pn_combine<<<CEILDIV((size_t)B_ * 32 * P_, 256), 256, 0, stream>>>(sc1, mean1, mask, fts1, 32);

  // -------- KNN2 (WMMA Gram tiles + async LDS double-buffer) --------
  pn_knn2prep<<<CEILDIV(B_ * P_, 256), 256, 0, stream>>>(fts1, mask, Xt2, nrm2);
  pn_knn_wmma<<<dim3(P_ / 16, B_), 32, 0, stream>>>(Xt2, nrm2, idx2);

  // -------- EdgeConv 2 (chunked over batches, bf16 edge-major) --------
  for (int c0 = 0; c0 < B_; c0 += CB_) {
    pn_edge_feat16<<<CEILDIV((size_t)CB_ * PK_ * 64, 256), 256, 0, stream>>>(
        fts1 + (size_t)c0 * 32 * P_, idx2 + (size_t)c0 * P_ * KNN_, ef16, 32, 64, CB_);
    pn_wmma_conv16<4, 2, 64><<<dim3(PK_ / 16, CB_), 32, 0, stream>>>(
        awE2W1, ef16, bufA16, ec2_bn1, PK_);
    pn_wmma_conv16<4, 2, 64><<<dim3(PK_ / 16, CB_), 32, 0, stream>>>(
        awE2W2, bufA16, bufB16, ec2_bn2, PK_);
    pn_wmma_conv16<4, 2, 64><<<dim3(PK_ / 16, CB_), 32, 0, stream>>>(
        awE2W3, bufB16, bufA16, ec2_bn3, PK_);
    pn_mean16<<<CEILDIV((size_t)CB_ * 64 * P_, 256), 256, 0, stream>>>(
        bufA16, mean2 + (size_t)c0 * 64 * P_, 64, CB_);
  }
  pn_wmma_conv<4, 1, 0, 0><<<dim3(P_ / 16, B_), 32, 0, stream>>>(
      awE2SC, fts1, sc2, ec2_scbn, nullptr, 32, 64, P_);
  pn_combine<<<CEILDIV((size_t)B_ * 64 * P_, 256), 256, 0, stream>>>(sc2, mean2, mask, fts2, 64);

  // -------- fusion + head --------
  pn_concat<<<CEILDIV((size_t)B_ * 96 * P_, 256), 256, 0, stream>>>(fts1, fts2, xcat);
  pn_wmma_conv<8, 3, 1, 1><<<dim3(P_ / 16, B_), 32, 0, stream>>>(
      awFUS, xcat, fused, fus_bn, mask, 96, 128, P_);
  pn_pool<<<B_, 128, 0, stream>>>(fused, counts, pooled);
  pn_fc<<<B_, 128, 0, stream>>>(pooled, fc1_w, fc1_b, fc2_w, fc2_b, out);
}
